// WeightAssigner_61727269978457
// MI455X (gfx1250) — matrix operand order repaired and checked by reading, not compiled, not run on hardware
//
#include <hip/hip_runtime.h>
#include <hip/hip_bf16.h>

// ---------------------------------------------------------------------------
// WeightAssigner pipeline for MI455X (gfx1250, wave32, WMMA).
// GEMMs: split-bf16 (hi + lo correction) via v_wmma_f32_16x16x32_bf16 with
// fp32 accumulation. A and B operands are PRE-PACKED in global memory in the
// exact WMMA lane layout, so fragments load as contiguous 32B global_load_b128
// pairs -- no LDS staging / swizzling inside the GEMM.
// GCN scatter-add is rewritten as a gather (deg == k+1 is constant).
// Edge top-k stages its 4KB score row via global_load_async_to_lds_b128.
// ---------------------------------------------------------------------------

typedef __attribute__((ext_vector_type(16))) __bf16 v16bf;
typedef __attribute__((ext_vector_type(8)))  float  v8f;

static constexpr int BATCH = 16;
static constexpr int NNODE = 1024;
static constexpr int HID   = 256;

// ---------------- packed-fragment index helpers -----------------------------
// A (16x32 bf16 tile, MxK): lane = (m%16) + 16*((k>>3)&1), e = 8*((k>>4)&1) + (k&7)
__device__ __forceinline__ size_t a_pack_idx(int row, int k, int K) {
  int mt = row >> 4, kt = k >> 5;
  int lane = (row & 15) + (((k >> 3) & 1) << 4);
  int e = (((k >> 4) & 1) << 3) + (k & 7);
  return ((((size_t)mt * (K >> 5) + kt) << 9) + (lane << 4)) + e;
}
// B (32x16 bf16 tile, KxN): lane = (n%16) + 16*((k>>3)&1), e = 8*((k>>4)&1) + (k&7)
__device__ __forceinline__ size_t b_pack_idx(int k, int n, int N) {
  int kt = k >> 5, nt = n >> 4;
  int lane = (n & 15) + (((k >> 3) & 1) << 4);
  int e = (((k >> 4) & 1) << 3) + (k & 7);
  return ((((size_t)kt * (N >> 4) + nt) << 9) + (lane << 4)) + e;
}

// ---------------- RNG helpers (deterministic hash-based Gumbel) -------------
__device__ __forceinline__ unsigned pcg_hash(unsigned x) {
  x = x * 747796405u + 2891336453u;
  unsigned w = ((x >> ((x >> 28u) + 4u)) ^ x) * 277803737u;
  return (w >> 22u) ^ w;
}
__device__ __forceinline__ float gumbel_h(unsigned a, unsigned b, unsigned c) {
  unsigned h = pcg_hash(a ^ pcg_hash(b ^ pcg_hash(c)));
  float u = (float)(h >> 8) * (1.0f / 16777216.0f) + 1e-7f;
  return -__logf(-__logf(u));
}

// ---------------- sort rows of x (bitonic, 1024) + packed bf16 split --------
__global__ void k_sort_split(const float* __restrict__ x,
                             __bf16* __restrict__ hi, __bf16* __restrict__ lo) {
  __shared__ float buf[1024];
  const int row = blockIdx.x;
  const float* xr = x + (size_t)row * 1024;
  for (int i = threadIdx.x; i < 1024; i += 256) buf[i] = xr[i];
  __syncthreads();
  for (int k = 2; k <= 1024; k <<= 1) {
    for (int j = k >> 1; j > 0; j >>= 1) {
      for (int i = threadIdx.x; i < 1024; i += 256) {
        int ixj = i ^ j;
        if (ixj > i) {
          float a = buf[i], b = buf[ixj];
          bool up = ((i & k) == 0);
          if ((a > b) == up) { buf[i] = b; buf[ixj] = a; }
        }
      }
      __syncthreads();
    }
  }
  for (int i = threadIdx.x; i < 1024; i += 256) {
    float v = buf[i];
    __bf16 h = (__bf16)v;
    size_t o = a_pack_idx(row, i, 1024);
    hi[o] = h;
    lo[o] = (__bf16)(v - (float)h);
  }
}

// ---------------- weight pack: fp32 [K,N] -> packed (hi, lo) bf16 -----------
__global__ void k_pack_w(const float* __restrict__ in,
                         __bf16* __restrict__ hi, __bf16* __restrict__ lo, int K, int N) {
  int i = blockIdx.x * 256 + threadIdx.x;
  if (i < K * N) {
    int k = i / N, n = i % N;
    float v = in[i];
    __bf16 h = (__bf16)v;
    size_t o = b_pack_idx(k, n, N);
    hi[o] = h;
    lo[o] = (__bf16)(v - (float)h);
  }
}

// ---------------- split-bf16 GEMM from packed operands ----------------------
// 256 threads = 8 wave32; block computes 128x64 C tile; wave w owns 16 rows.
// Fragments load as contiguous 32B (2x global_load_b128) per operand.
__global__ void k_gemm(const __bf16* __restrict__ Ah, const __bf16* __restrict__ Al,
                       const __bf16* __restrict__ Bh, const __bf16* __restrict__ Bl,
                       float* __restrict__ C, const float* __restrict__ bias,
                       int M, int N, int K,
                       __bf16* __restrict__ Cph, __bf16* __restrict__ Cpl) {
  const int tiles_n = N >> 6;
  const int bm = blockIdx.x / tiles_n, bn = blockIdx.x % tiles_n;
  const int tid = threadIdx.x;
  const int wave = tid >> 5, lane = tid & 31;
  const int mt = (bm << 3) + wave;        // global 16-row tile id
  const int row0 = mt << 4;
  const int col0 = bn << 6;
  const int KT = K >> 5, NT16 = N >> 4;

  v8f acc[4];
#pragma unroll
  for (int nt = 0; nt < 4; nt++)
#pragma unroll
    for (int e = 0; e < 8; e++) acc[nt][e] = 0.0f;

  const __bf16* aph = Ah + (((size_t)mt * KT) << 9) + (lane << 4);
  const __bf16* apl = Al + (((size_t)mt * KT) << 9) + (lane << 4);

  for (int kt = 0; kt < KT; ++kt) {
    v16bf ah = *(const v16bf*)(aph + ((size_t)kt << 9));
    v16bf al = *(const v16bf*)(apl + ((size_t)kt << 9));
    if (kt + 1 < KT)
      __builtin_prefetch(aph + ((size_t)(kt + 1) << 9), 0, 1);
    size_t bbase = (((size_t)kt * NT16 + (col0 >> 4)) << 9) + (lane << 4);
#pragma unroll
    for (int nt = 0; nt < 4; nt++) {
      size_t o = bbase + ((size_t)nt << 9);
      v16bf bh = *(const v16bf*)(Bh + o);
      v16bf bl = *(const v16bf*)(Bl + o);
      acc[nt] = __builtin_amdgcn_wmma_f32_16x16x32_bf16(false, ah, false, bh,
                                                        (short)0, acc[nt], false, false);
      acc[nt] = __builtin_amdgcn_wmma_f32_16x16x32_bf16(false, ah, false, bl,
                                                        (short)0, acc[nt], false, false);
      acc[nt] = __builtin_amdgcn_wmma_f32_16x16x32_bf16(false, al, false, bh,
                                                        (short)0, acc[nt], false, false);
    }
  }

#pragma unroll
  for (int nt = 0; nt < 4; nt++) {
    int col = col0 + (nt << 4) + (lane & 15);
#pragma unroll
    for (int r = 0; r < 8; r++) {
      int row = row0 + r + ((lane >= 16) ? 8 : 0);
      float v = acc[nt][r];
      if (bias) v += bias[col];
      C[(size_t)row * N + col] = v;
      if (Cph) {  // pack for the next GEMM's A operand (K-dim == N here)
        __bf16 h = (__bf16)v;
        size_t o = a_pack_idx(row, col, N);
        Cph[o] = h;
        Cpl[o] = (__bf16)(v - (float)h);
      }
    }
  }
}

// ---------------- sel init --------------------------------------------------
__global__ void k_init_sel(int* __restrict__ sel) {
  int b = blockIdx.x;
  for (int i = threadIdx.x; i < NNODE; i += 256) sel[b * NNODE + i] = i;
}

// ---------------- per-row Gumbel-top-k over edge logits ---------------------
// Stages the full 4KB edge_score row into LDS with async-to-LDS loads.
__global__ void k_edge_topk(const float* __restrict__ edge_score, const int* __restrict__ sel,
                            int n_s, int kk, int layer, int* __restrict__ idx_out) {
  __shared__ float row_s[1024];
  __shared__ float val[1024];
  __shared__ float rv[256];
  __shared__ int   ri[256];
  const int b = blockIdx.x / n_s;
  const int i = blockIdx.x % n_s;
  const int tid = threadIdx.x;
  const int si = sel[b * NNODE + i];

  // async copy: 256 lanes x 16B = 4KB = one full original score row
  {
    const float* grow = edge_score + ((size_t)b * NNODE + si) * NNODE;
    unsigned long long ga = (unsigned long long)(grow) + (unsigned long long)tid * 16ull;
    unsigned lo = (unsigned)(size_t)(&row_s[0]) + (unsigned)tid * 16u;  // LDS byte offset
    asm volatile("global_load_async_to_lds_b128 %0, %1, off"
                 :: "v"(lo), "v"(ga) : "memory");
    asm volatile("s_wait_asynccnt 0" ::: "memory");
  }
  __syncthreads();

  for (int j = tid; j < n_s; j += 256) {
    int sj = sel[b * NNODE + j];
    float l;
    if (si == sj) l = -1e30f;
    else l = -row_s[sj] +
             gumbel_h(0x9E3779B9u * (unsigned)(layer + 1), (unsigned)(b * n_s + i), (unsigned)j);
    val[j] = l;
  }
  __syncthreads();
  for (int t = 0; t < kk; t++) {
    float bv = -1e38f; int bi = 0;
    for (int j = tid; j < n_s; j += 256) {
      float v = val[j];
      if (v > bv) { bv = v; bi = j; }
    }
    rv[tid] = bv; ri[tid] = bi;
    __syncthreads();
    for (int s = 128; s > 0; s >>= 1) {
      if (tid < s && rv[tid + s] > rv[tid]) { rv[tid] = rv[tid + s]; ri[tid] = ri[tid + s]; }
      __syncthreads();
    }
    if (tid == 0) {
      idx_out[((size_t)b * n_s + i) * kk + t] = ri[0];
      val[ri[0]] = -1e38f;
    }
    __syncthreads();
  }
}

// ---------------- node Gumbel-top-(n/2) via bitonic sort --------------------
__global__ void k_node_topk(const float* __restrict__ node_score, const int* __restrict__ sel_in,
                            int n_s, int layer, int* __restrict__ nidx_out, int* __restrict__ sel_out) {
  __shared__ float val[1024];
  __shared__ int   idx[1024];
  const int b = blockIdx.x;
  const int tid = threadIdx.x;
  for (int i = tid; i < n_s; i += 256) {
    int si = sel_in[b * NNODE + i];
    val[i] = -node_score[b * NNODE + si] +
             gumbel_h(0xA5A50000u + (unsigned)layer, (unsigned)b, (unsigned)i);
    idx[i] = i;
  }
  __syncthreads();
  for (int k = 2; k <= n_s; k <<= 1) {
    for (int j = k >> 1; j > 0; j >>= 1) {
      for (int i = tid; i < n_s; i += 256) {
        int ixj = i ^ j;
        if (ixj > i) {
          bool descFirst = ((i & k) == 0);
          float a = val[i], c = val[ixj];
          if ((a < c) == descFirst) {
            val[i] = c; val[ixj] = a;
            int t = idx[i]; idx[i] = idx[ixj]; idx[ixj] = t;
          }
        }
      }
      __syncthreads();
    }
  }
  int n_new = n_s >> 1;
  for (int i = tid; i < n_new; i += 256) {
    int li = idx[i];
    nidx_out[b * n_new + i] = li;
    sel_out[b * NNODE + i] = sel_in[b * NNODE + li];
  }
}

// ---------------- GCN aggregate (gather form) + bias + LeakyReLU ------------
__global__ void k_gcn_agg(const float* __restrict__ h, const int* __restrict__ idx,
                          const float* __restrict__ bias, int n_s, int kk,
                          float* __restrict__ out) {
  const int node = blockIdx.x;            // b*n_s + i
  const int b = node / n_s, i = node % n_s;
  const int d = threadIdx.x;              // HID = 256
  const float norm = 1.0f / (float)(kk + 1);
  const float* hb = h + (size_t)b * n_s * HID;
  float acc = hb[(size_t)i * HID + d];
  const int* row = idx + ((size_t)b * n_s + i) * kk;
  for (int j = 0; j < kk; j++) acc += hb[(size_t)row[j] * HID + d];
  acc = acc * norm + bias[d];
  acc = acc > 0.0f ? acc : 0.01f * acc;
  out[(size_t)node * HID + d] = acc;
}

// ---------------- node feature gather -> packed A for next GEMM -------------
__global__ void k_gather_pack(const float* __restrict__ in, const int* __restrict__ nidx,
                              int n_old, int n_new,
                              __bf16* __restrict__ oh, __bf16* __restrict__ ol) {
  const int node = blockIdx.x;            // b*n_new + i
  const int b = node / n_new, i = node % n_new;
  const int d = threadIdx.x;
  int src = nidx[b * n_new + i];
  float v = in[((size_t)b * n_old + src) * HID + d];
  size_t o = a_pack_idx(node, d, HID);
  __bf16 hv = (__bf16)v;
  oh[o] = hv;
  ol[o] = (__bf16)(v - (float)hv);
}

// ---------------- final 256-dot + pairwise sigmoid --------------------------
__global__ void k_final_score(const float* __restrict__ x, const float* __restrict__ W2,
                              const float* __restrict__ b2, float* __restrict__ s) {
  __shared__ float red[256];
  int node = blockIdx.x, d = threadIdx.x;
  red[d] = x[(size_t)node * HID + d] * W2[d];
  __syncthreads();
  for (int st = 128; st > 0; st >>= 1) {
    if (d < st) red[d] += red[d + st];
    __syncthreads();
  }
  if (d == 0) s[node] = red[0] + b2[0];
}

__global__ void k_pairwise(const float* __restrict__ s, float* __restrict__ out, int n_f) {
  int b = blockIdx.y, i = blockIdx.x, j = threadIdx.x;
  float d = s[b * n_f + i] - s[b * n_f + j];
  out[((size_t)b * n_f + i) * n_f + j] = 1.0f / (1.0f + __expf(-d));
}

__global__ void k_write_selected(const int* __restrict__ sel, float* __restrict__ out, int n_f) {
  int b = blockIdx.x, i = threadIdx.x;
  out[b * n_f + i] = (float)sel[b * NNODE + i];
}

// ---------------------------------------------------------------------------
extern "C" void kernel_launch(void* const* d_in, const int* in_sizes, int n_in,
                              void* d_out, int out_size, void* d_ws, size_t ws_size,
                              hipStream_t stream) {
  (void)in_sizes; (void)n_in; (void)out_size; (void)ws_size;
  const float* x   = (const float*)d_in[0];   // [16,1024,1024]
  const float* es  = (const float*)d_in[1];   // [16,1024,1024]
  const float* nsc = (const float*)d_in[2];   // [16,1024]
  const float* W1  = (const float*)d_in[3];   // [1024,256]
  const float* b1  = (const float*)d_in[4];   // [256]
  const float* gW  = (const float*)d_in[5];   // [3,256,256]
  const float* gb  = (const float*)d_in[6];   // [3,256]
  const float* W2  = (const float*)d_in[7];   // [256]
  const float* b2  = (const float*)d_in[8];   // [1]
  float* out = (float*)d_out;

  char* ws = (char*)d_ws;
  size_t off = 0;
  auto alloc = [&](size_t bytes) -> void* {
    void* p = ws + off;
    off += (bytes + 255) & ~(size_t)255;
    return p;
  };

  const int M0 = BATCH * NNODE;             // 16384
  __bf16* Ahi = (__bf16*)alloc((size_t)M0 * 1024 * 2);   // packed sorted x
  __bf16* Alo = (__bf16*)alloc((size_t)M0 * 1024 * 2);
  __bf16* W1h = (__bf16*)alloc(1024 * 256 * 2);          // packed weights
  __bf16* W1l = (__bf16*)alloc(1024 * 256 * 2);
  __bf16* Gh  = (__bf16*)alloc(3 * 256 * 256 * 2);
  __bf16* Gl  = (__bf16*)alloc(3 * 256 * 256 * 2);
  float*  hb  = (float*)alloc((size_t)M0 * HID * 4);     // GEMM f32 out
  float*  xbf = (float*)alloc((size_t)M0 * HID * 4);     // aggregate f32 out
  __bf16* xph = (__bf16*)alloc((size_t)M0 * HID * 2);    // packed A for layer GEMMs
  __bf16* xpl = (__bf16*)alloc((size_t)M0 * HID * 2);
  int* sel0  = (int*)alloc(BATCH * NNODE * 4);
  int* sel1  = (int*)alloc(BATCH * NNODE * 4);
  int* sel2  = (int*)alloc(BATCH * NNODE * 4);
  int* nidx1 = (int*)alloc(BATCH * 512 * 4);
  int* nidx2 = (int*)alloc(BATCH * 256 * 4);
  int* idx0  = (int*)alloc((size_t)BATCH * 1024 * 8 * 4);
  int* idx1  = (int*)alloc((size_t)BATCH * 512 * 4 * 4);
  int* idx2  = (int*)alloc((size_t)BATCH * 256 * 2 * 4);
  float* sbuf = (float*)alloc(BATCH * 256 * 4);

  // 1) sort rows of x + packed bf16 split
  k_sort_split<<<M0, 256, 0, stream>>>(x, Ahi, Alo);

  // 2) pack weights into WMMA B-fragment order
  k_pack_w<<<(1024 * 256 + 255) / 256, 256, 0, stream>>>(W1, W1h, W1l, 1024, 256);
  for (int l = 0; l < 3; l++)
    k_pack_w<<<(256 * 256 + 255) / 256, 256, 0, stream>>>(gW + (size_t)l * 256 * 256,
                                                          Gh + (size_t)l * 256 * 256,
                                                          Gl + (size_t)l * 256 * 256, 256, 256);

  // 3) sel init
  k_init_sel<<<BATCH, 256, 0, stream>>>(sel0);

  // 4) x = sort(x) @ W1 + b1  -> packed xp (A for layer-0 GEMM)
  k_gemm<<<(M0 / 128) * (256 / 64), 256, 0, stream>>>(Ahi, Alo, W1h, W1l,
                                                      hb, b1, M0, 256, 1024, xph, xpl);

  // ---- layer 0 (n=1024, k=8) ----
  k_edge_topk<<<BATCH * 1024, 256, 0, stream>>>(es, sel0, 1024, 8, 0, idx0);
  k_gemm<<<(M0 / 128) * (256 / 64), 256, 0, stream>>>(xph, xpl, Gh, Gl,
                                                      hb, (const float*)nullptr, M0, 256, 256,
                                                      (__bf16*)nullptr, (__bf16*)nullptr);
  k_gcn_agg<<<M0, 256, 0, stream>>>(hb, idx0, gb + 0 * 256, 1024, 8, xbf);
  k_node_topk<<<BATCH, 256, 0, stream>>>(nsc, sel0, 1024, 1, nidx1, sel1);

  // ---- layer 1 (n=512, k=4) ----
  const int M1 = BATCH * 512;
  k_gather_pack<<<M1, 256, 0, stream>>>(xbf, nidx1, 1024, 512, xph, xpl);
  k_edge_topk<<<BATCH * 512, 256, 0, stream>>>(es, sel1, 512, 4, 1, idx1);
  k_gemm<<<(M1 / 128) * (256 / 64), 256, 0, stream>>>(xph, xpl,
                                                      Gh + 1 * 256 * 256, Gl + 1 * 256 * 256,
                                                      hb, (const float*)nullptr, M1, 256, 256,
                                                      (__bf16*)nullptr, (__bf16*)nullptr);
  k_gcn_agg<<<M1, 256, 0, stream>>>(hb, idx1, gb + 1 * 256, 512, 4, xbf);
  k_node_topk<<<BATCH, 256, 0, stream>>>(nsc, sel1, 512, 2, nidx2, sel2);

  // ---- layer 2 (n=256, k=2) ----
  const int M2 = BATCH * 256;
  k_gather_pack<<<M2, 256, 0, stream>>>(xbf, nidx2, 512, 256, xph, xpl);
  k_edge_topk<<<BATCH * 256, 256, 0, stream>>>(es, sel2, 256, 2, 2, idx2);
  k_gemm<<<(M2 / 128) * (256 / 64), 256, 0, stream>>>(xph, xpl,
                                                      Gh + 2 * 256 * 256, Gl + 2 * 256 * 256,
                                                      hb, (const float*)nullptr, M2, 256, 256,
                                                      (__bf16*)nullptr, (__bf16*)nullptr);
  k_gcn_agg<<<M2, 256, 0, stream>>>(hb, idx2, gb + 2 * 256, 256, 2, xbf);

  // ---- final head ----
  k_final_score<<<M2, 256, 0, stream>>>(xbf, W2, b2, sbuf);
  {
    dim3 grid(256, BATCH);
    k_pairwise<<<grid, 256, 0, stream>>>(sbuf, out, 256);
  }
  k_write_selected<<<BATCH, 256, 0, stream>>>(sel2, out + (size_t)BATCH * 256 * 256, 256);
}
